// SpatialAttention_4234837754559
// MI455X (gfx1250) — compile-verified, matrix operand order checked by reading, not run on hardware
//
#include <hip/hip_runtime.h>
#include <math.h>

// ---------------------------------------------------------------------------
// Types for CDNA5 WMMA (wave32)
// ---------------------------------------------------------------------------
typedef __attribute__((ext_vector_type(8)))  __bf16 v8bf;
typedef __attribute__((ext_vector_type(16))) __bf16 v16bf;
typedef __attribute__((ext_vector_type(8)))  float  v8f;

union FragBF { v16bf v; v8bf h[2]; };

// D = A(16x32 bf16) * B(32x16 bf16) + C(16x16 f32)
__device__ __forceinline__ v8f wmma_bf16(v16bf a, v16bf b, v8f c) {
    return __builtin_amdgcn_wmma_f32_16x16x32_bf16(
        /*neg_a=*/false, a, /*neg_b=*/false, b,
        /*c_mod=*/(short)0, c, /*reuse_a=*/false, /*reuse_b=*/false);
}

// Load a 16(row) x 32(K) bf16 fragment from a row-major (ld = K-stride) tile.
// ISA 16-bit A/B layout (wave32): lanes 0-15 hold row = lane, K = {0..7,16..23};
// lanes 16-31 hold row = lane-16, K = {8..15,24..31}.  Two 16B loads per lane.
__device__ __forceinline__ v16bf load_frag(const __bf16* __restrict__ base,
                                           int ld, int k0) {
    int lane = threadIdx.x & 31;
    int half = lane >> 4;
    int r    = lane & 15;
    const __bf16* p = base + (size_t)r * ld + k0 + half * 8;
    FragBF f;
    f.h[0] = *reinterpret_cast<const v8bf*>(p);
    f.h[1] = *reinterpret_cast<const v8bf*>(p + 16);
    return f.v;
}

// Async copy of 16 bytes per lane: global -> LDS, tracked by ASYNCcnt.
// Generic shared-memory pointers carry the LDS offset in their low 32 bits
// (flat LDS aperture).
__device__ __forceinline__ void async_copy16(const __bf16* g, __bf16* l) {
    unsigned ldsAddr = (unsigned)(size_t)l;
    asm volatile("global_load_async_to_lds_b128 %0, %1, off"
                 :: "v"(ldsAddr), "v"((const void*)g)
                 : "memory");
}
__device__ __forceinline__ void wait_async_all() {
    asm volatile("s_wait_asynccnt 0x0" ::: "memory");
}

// ---------------------------------------------------------------------------
// Problem constants (from reference)
// ---------------------------------------------------------------------------
#define HDIM   1024
#define SEQ    8192
#define NHEAD  16
#define DHEAD  64
#define FRAMES 8
#define SPAT   1024      // 32*32 tokens per frame
#define NQKV   3072

// LDS row paddings (bf16 elements) to rotate bank mapping (64 banks x 4B)
#define LDA_A  40        // 32-col tiles  (80B rows, 16B aligned)
#define LDA_K  72        // 64-col tiles  (144B rows, 16B aligned)

// ---------------------------------------------------------------------------
// Kernel 1: LayerNorm (f32 in) -> bf16 out.  One 256-thread block per token.
// ---------------------------------------------------------------------------
__global__ void ln_kernel(const float* __restrict__ x,
                          const float* __restrict__ gamma,
                          const float* __restrict__ beta,
                          __bf16* __restrict__ xn) {
    int t = blockIdx.x;
    const float* row = x + (size_t)t * HDIM;
    float s = 0.f, ss = 0.f;
    for (int i = threadIdx.x; i < HDIM; i += 256) {
        float v = row[i];
        s += v; ss += v * v;
    }
    #pragma unroll
    for (int m = 16; m >= 1; m >>= 1) {
        s  += __shfl_xor(s,  m, 32);
        ss += __shfl_xor(ss, m, 32);
    }
    __shared__ float red0[8], red1[8];
    int wid = threadIdx.x >> 5, lid = threadIdx.x & 31;
    if (lid == 0) { red0[wid] = s; red1[wid] = ss; }
    __syncthreads();
    s = 0.f; ss = 0.f;
    #pragma unroll
    for (int i = 0; i < 8; ++i) { s += red0[i]; ss += red1[i]; }
    float mu   = s * (1.0f / HDIM);
    float var  = ss * (1.0f / HDIM) - mu * mu;
    float rstd = rsqrtf(var + 1e-5f);
    __bf16* orow = xn + (size_t)t * HDIM;
    for (int i = threadIdx.x; i < HDIM; i += 256) {
        float v = (row[i] - mu) * rstd * gamma[i] + beta[i];
        orow[i] = (__bf16)v;
    }
}

// ---------------------------------------------------------------------------
// Kernel 2: f32 -> bf16 convert (weights)
// ---------------------------------------------------------------------------
__global__ void cvt_bf16_kernel(const float* __restrict__ in,
                                __bf16* __restrict__ out, int n) {
    int i = blockIdx.x * 256 + threadIdx.x;
    if (i < n) out[i] = (__bf16)in[i];
}

// ---------------------------------------------------------------------------
// Kernel 3: generic bf16 GEMM  C[M,N] = A[M,K] * B[N,K]^T
// 4 waves / block; wave tile 64(M) x 64(N); block tile 64 x 256.
// A K-chunk (64x32) staged in LDS via async-to-LDS (double-buffered), shared
// by all 4 waves; 16 wmma per chunk per wave, 8 global b128 for B.
// __launch_bounds__(128, 1): ~190 VGPRs live (128 accumulator) -- raise the
// per-wave VGPR budget so the accumulator tile never spills to scratch.
// ---------------------------------------------------------------------------
__global__ void __launch_bounds__(128, 1)
gemm_bf16_kernel(const __bf16* __restrict__ A,
                 const __bf16* __restrict__ B,
                 float* __restrict__ Cf,
                 __bf16* __restrict__ Cb,
                 int M, int N, int K) {
    __shared__ __align__(16) __bf16 asmem[2][64 * LDA_A];   // 2 x 5KB A chunks
    int tid  = threadIdx.x;
    int wave = tid >> 5, lane = tid & 31;
    int half = lane >> 4, lcol = lane & 15;
    int m0 = blockIdx.x * 64;
    int n0 = blockIdx.y * 256 + wave * 64;
    const __bf16* Abase = A + (size_t)m0 * K;

    // stage a 64x32 A chunk: 256 x 16B segments over 128 threads
    auto stage_a = [&](int k0, int buf) {
        #pragma unroll
        for (int s = 0; s < 2; ++s) {
            int seg = tid + s * 128;
            int row = seg >> 2, sg = seg & 3;
            async_copy16(Abase + (size_t)row * K + k0 + sg * 8,
                         &asmem[buf][row * LDA_A + sg * 8]);
        }
    };

    v8f acc[4][4] = {};

    auto compute = [&](int k0, int buf) {
        v16bf a[4];
        #pragma unroll
        for (int mt = 0; mt < 4; ++mt)
            a[mt] = load_frag(&asmem[buf][mt * 16 * LDA_A], LDA_A, 0);
        #pragma unroll
        for (int nt = 0; nt < 4; ++nt) {
            v16bf b = load_frag(B + (size_t)(n0 + nt * 16) * K, K, k0);
            #pragma unroll
            for (int mt = 0; mt < 4; ++mt)
                acc[mt][nt] = wmma_bf16(a[mt], b, acc[mt][nt]);
        }
    };

    stage_a(0, 0);
    wait_async_all();
    __syncthreads();

    int k0 = 0;
    for (; k0 < K - 32; k0 += 32) {             // steady state: branch-free
        int cur = (k0 >> 5) & 1;
        stage_a(k0 + 32, cur ^ 1);
        #pragma unroll
        for (int nt = 0; nt < 4; ++nt)          // prefetch next B chunk
            __builtin_prefetch(B + (size_t)(n0 + nt * 16) * K + k0 + 32, 0, 1);
        compute(k0, cur);
        wait_async_all();
        __syncthreads();
    }
    compute(k0, (k0 >> 5) & 1);                 // peeled last chunk

    #pragma unroll
    for (int mt = 0; mt < 4; ++mt) {
        #pragma unroll
        for (int nt = 0; nt < 4; ++nt) {
            #pragma unroll
            for (int c = 0; c < 8; ++c) {
                int row = m0 + mt * 16 + c + 8 * half;  // C layout
                int n   = n0 + nt * 16 + lcol;
                if (Cf) Cf[(size_t)row * N + n] = acc[mt][nt][c];
                else    Cb[(size_t)row * N + n] = (__bf16)acc[mt][nt][c];
            }
        }
    }
}

// ---------------------------------------------------------------------------
// Kernel 4: 2D-RoPE on q,k + permute; v transposed.
// qkv: (8192, 3072) bf16 ->  Qr,Kr: (128 bh, 1024 s, 64 d)   Vt: (128, 64, 1024)
// ---------------------------------------------------------------------------
__global__ void rope_kernel(const __bf16* __restrict__ qkv,
                            __bf16* __restrict__ Qr,
                            __bf16* __restrict__ Kr,
                            __bf16* __restrict__ Vt) {
    int idx = blockIdx.x * 256 + threadIdx.x;      // 8192*1024 threads
    int t = idx >> 10;                             // token
    int c = idx & 1023;                            // head*64 + d
    int bf = t >> 10, s = t & 1023;
    int head = c >> 6, d = c & 63;
    int xpos = s & 31, ypos = s >> 5;              // width=32
    int j = d & 15;
    float pos  = (d < 32) ? (float)xpos : (float)ypos;
    // inv_freq = 10000^(-j/16)
    float freq = __expf(-(float)j * (1.0f / 16.0f) * 9.210340371976184f);
    float ang  = pos * freq;
    float sn, cs;
    __sincosf(ang, &sn, &cs);
    const __bf16* row = qkv + (size_t)t * NQKV;
    float qv = (float)row[c];
    float kv = (float)row[HDIM + c];
    int   dp = (d < 32) ? (c + 32) : (c - 32);
    float qp = (float)row[dp];
    float kp = (float)row[HDIM + dp];
    float sgn = (d < 32) ? -1.0f : 1.0f;
    float qo = qv * cs + sgn * qp * sn;
    float ko = kv * cs + sgn * kp * sn;
    size_t bh = (size_t)(bf * NHEAD + head);
    Qr[bh * SPAT * DHEAD + (size_t)s * DHEAD + d] = (__bf16)qo;
    Kr[bh * SPAT * DHEAD + (size_t)s * DHEAD + d] = (__bf16)ko;
    Vt[bh * DHEAD * SPAT + (size_t)d * SPAT + s] = (__bf16)(float)row[2 * HDIM + c];
}

// ---------------------------------------------------------------------------
// Kernel 5: flash-style attention.  grid = (16 qblocks, 128 bh), block = 128.
// Each wave owns 16 q rows.  K-block (32x64) and V-block (64x32) staged in
// LDS (async-to-LDS, double-buffered) shared by all 4 waves.  Last key block
// peeled so the steady-state body is branch-free.
// ---------------------------------------------------------------------------
__global__ void attn_kernel(const __bf16* __restrict__ Qr,
                            const __bf16* __restrict__ Kr,
                            const __bf16* __restrict__ Vt,
                            __bf16* __restrict__ AO) {
    __shared__ __align__(16) __bf16 ksmem[2][32 * LDA_K];  // 32 keys x 64 d (padded)
    __shared__ __align__(16) __bf16 vsmem[2][64 * LDA_A];  // 64 d x 32 seq (padded)
    __shared__ __align__(16) __bf16 pbuf[4][16 * LDA_A];   // per-wave P reshape

    int bh   = blockIdx.y;
    int tid  = threadIdx.x;
    int wave = tid >> 5;
    int lane = tid & 31;
    int half = lane >> 4, lcol = lane & 15;
    int q0   = blockIdx.x * 64 + wave * 16;

    const __bf16* Qb = Qr + (size_t)bh * SPAT * DHEAD + (size_t)q0 * DHEAD;
    const __bf16* Kb = Kr + (size_t)bh * SPAT * DHEAD;
    const __bf16* Vb = Vt + (size_t)bh * DHEAD * SPAT;

    // stage K/V tiles for key block kb into buffer `buf`
    auto stage_kv = [&](int kb, int buf) {
        #pragma unroll
        for (int s = 0; s < 2; ++s) {           // K: 32 rows x 8 x 16B
            int seg = tid + s * 128;
            int row = seg >> 3, sg = seg & 7;
            async_copy16(Kb + (size_t)(kb + row) * DHEAD + sg * 8,
                         &ksmem[buf][row * LDA_K + sg * 8]);
        }
        #pragma unroll
        for (int s = 0; s < 2; ++s) {           // V: 64 rows x 4 x 16B
            int seg = tid + s * 128;
            int row = seg >> 2, sg = seg & 3;
            async_copy16(Vb + (size_t)row * SPAT + kb + sg * 8,
                         &vsmem[buf][row * LDA_A + sg * 8]);
        }
    };

    // Q fragments (16 x 64 -> two K-chunks), resident for the whole loop
    v16bf qf0 = load_frag(Qb, DHEAD, 0);
    v16bf qf1 = load_frag(Qb, DHEAD, 32);

    v8f o[4] = {};            // 16 x 64 f32 output accumulator
    float mi[8], li[8];
    #pragma unroll
    for (int c = 0; c < 8; ++c) { mi[c] = -1e30f; li[c] = 0.f; }

    const float scale = 0.125f;                 // 64^-0.5

    auto process = [&](int buf) {
        // ---- S = Q K^T for 32 keys (two 16-wide N tiles), from LDS ----
        v8f s0 = {}, s1 = {};
        s0 = wmma_bf16(qf0, load_frag(&ksmem[buf][0],            LDA_K, 0),  s0);
        s0 = wmma_bf16(qf1, load_frag(&ksmem[buf][0],            LDA_K, 32), s0);
        s1 = wmma_bf16(qf0, load_frag(&ksmem[buf][16 * LDA_K],   LDA_K, 0),  s1);
        s1 = wmma_bf16(qf1, load_frag(&ksmem[buf][16 * LDA_K],   LDA_K, 32), s1);

        // ---- online softmax (rows live within 16-lane halves) ----
        float cf[8];
        #pragma unroll
        for (int c = 0; c < 8; ++c) {
            float v0 = s0[c] * scale;
            float v1 = s1[c] * scale;
            float mx = fmaxf(v0, v1);
            #pragma unroll
            for (int m = 8; m >= 1; m >>= 1)
                mx = fmaxf(mx, __shfl_xor(mx, m, 32));
            float mnew = fmaxf(mi[c], mx);
            float corr = __expf(mi[c] - mnew);
            float p0 = __expf(v0 - mnew);
            float p1 = __expf(v1 - mnew);
            float rs = p0 + p1;
            #pragma unroll
            for (int m = 8; m >= 1; m >>= 1)
                rs += __shfl_xor(rs, m, 32);
            li[c] = li[c] * corr + rs;
            mi[c] = mnew;
            cf[c] = corr;
            s0[c] = p0;
            s1[c] = p1;
        }
        #pragma unroll
        for (int dt = 0; dt < 4; ++dt)
            #pragma unroll
            for (int c = 0; c < 8; ++c)
                o[dt][c] *= cf[c];

        // ---- reshape P (C-layout) -> A-frag layout through LDS ----
        __bf16* pb = &pbuf[wave][0];
        #pragma unroll
        for (int c = 0; c < 8; ++c) {
            int row = c + 8 * half;
            pb[row * LDA_A + lcol]      = (__bf16)s0[c];
            pb[row * LDA_A + 16 + lcol] = (__bf16)s1[c];
        }
        __builtin_amdgcn_wave_barrier();   // same-wave LDS is in-order
        v16bf pf = load_frag(pb, LDA_A, 0);
        __builtin_amdgcn_wave_barrier();

        // ---- O += P * V  (V tile from LDS, rows = d, contiguous in seq) ----
        #pragma unroll
        for (int dt = 0; dt < 4; ++dt) {
            v16bf bv = load_frag(&vsmem[buf][dt * 16 * LDA_A], LDA_A, 0);
            o[dt] = wmma_bf16(pf, bv, o[dt]);
        }
    };

    stage_kv(0, 0);
    wait_async_all();
    __syncthreads();

    int kb = 0;
    for (; kb < SPAT - 32; kb += 32) {          // steady state: branch-free
        int cur = (kb >> 5) & 1;
        stage_kv(kb + 32, cur ^ 1);
        process(cur);
        wait_async_all();
        __syncthreads();
    }
    process((kb >> 5) & 1);                     // peeled last block

    // ---- normalize and write (bh,s,d) back into token-major (8192,1024) ----
    int bfi = bh >> 4, hd = bh & 15;
    #pragma unroll
    for (int dt = 0; dt < 4; ++dt) {
        #pragma unroll
        for (int c = 0; c < 8; ++c) {
            int srow  = q0 + c + 8 * half;
            int dcol  = dt * 16 + lcol;
            size_t tok = (size_t)bfi * SPAT + srow;
            AO[tok * HDIM + hd * DHEAD + dcol] = (__bf16)(o[dt][c] / li[c]);
        }
    }
}

// ---------------------------------------------------------------------------
// Host-side launch
// ---------------------------------------------------------------------------
extern "C" void kernel_launch(void* const* d_in, const int* in_sizes, int n_in,
                              void* d_out, int out_size, void* d_ws, size_t ws_size,
                              hipStream_t stream) {
    const float* x      = (const float*)d_in[0];
    const float* w_qkv  = (const float*)d_in[1];
    const float* w_out  = (const float*)d_in[2];
    const float* gamma  = (const float*)d_in[3];
    const float* beta   = (const float*)d_in[4];
    float* out = (float*)d_out;

    // workspace carve-up (bf16 scratch, ~136 MB)
    char* ws = (char*)d_ws;
    size_t off = 0;
    __bf16* XN   = (__bf16*)(ws + off); off += (size_t)SEQ * HDIM * 2;        // 16 MB
    __bf16* WQKV = (__bf16*)(ws + off); off += (size_t)NQKV * HDIM * 2;       //  6 MB
    __bf16* WOUT = (__bf16*)(ws + off); off += (size_t)HDIM * HDIM * 2;       //  2 MB
    __bf16* QKV  = (__bf16*)(ws + off); off += (size_t)SEQ * NQKV * 2;        // 48 MB
    __bf16* QR   = (__bf16*)(ws + off); off += (size_t)SEQ * HDIM * 2;        // 16 MB
    __bf16* KR   = (__bf16*)(ws + off); off += (size_t)SEQ * HDIM * 2;        // 16 MB
    __bf16* VT   = (__bf16*)(ws + off); off += (size_t)SEQ * HDIM * 2;        // 16 MB
    __bf16* AO   = (__bf16*)(ws + off); off += (size_t)SEQ * HDIM * 2;        // 16 MB
    (void)ws_size; (void)in_sizes; (void)n_in; (void)out_size;

    // 1. LayerNorm -> bf16
    ln_kernel<<<SEQ, 256, 0, stream>>>(x, gamma, beta, XN);

    // 2. weight converts
    cvt_bf16_kernel<<<(NQKV * HDIM) / 256, 256, 0, stream>>>(w_qkv, WQKV, NQKV * HDIM);
    cvt_bf16_kernel<<<(HDIM * HDIM) / 256, 256, 0, stream>>>(w_out, WOUT, HDIM * HDIM);

    // 3. QKV GEMM: (8192,1024) x (3072,1024)^T -> bf16 (8192,3072)
    gemm_bf16_kernel<<<dim3(SEQ / 64, NQKV / 256), 128, 0, stream>>>(
        XN, WQKV, nullptr, QKV, SEQ, NQKV, HDIM);

    // 4. RoPE + head permute + V transpose
    rope_kernel<<<(SEQ * HDIM) / 256, 256, 0, stream>>>(QKV, QR, KR, VT);

    // 5. attention: 128 (bh) x 16 q-blocks, 4 waves each
    attn_kernel<<<dim3(SPAT / 64, FRAMES * NHEAD), 128, 0, stream>>>(QR, KR, VT, AO);

    // 6. out projection: (8192,1024) x (1024,1024)^T -> f32 d_out
    gemm_bf16_kernel<<<dim3(SEQ / 64, HDIM / 256), 128, 0, stream>>>(
        AO, WOUT, out, nullptr, SEQ, HDIM, HDIM);
}
